// peepLSTM_71416716198166
// MI455X (gfx1250) — compile-verified, matrix operand order checked by reading, not run on hardware
//
#include <hip/hip_runtime.h>

// ---------------------------------------------------------------------------
// peephole LSTM for MI455X (gfx1250, wave32, WMMA bf16 16x16x32 + TDM)
//   SEQ=1024 (1023 steps), H=256, C=128, B=256
// ---------------------------------------------------------------------------

#define SEQ   1024
#define NSTEP 1023
#define Hd    256
#define Cd    128
#define Bd    256

// LDS row pitch for staged Wh: 256 elems + 16 elems pad (512B + 32B) -> 544B
// rotates banks by 8 per row so 16 lanes reading 16 rows don't conflict.
#define WH_PITCH 272

typedef __attribute__((ext_vector_type(16))) __bf16 v16bf;
typedef __attribute__((ext_vector_type(8)))  float  v8f;
typedef __attribute__((ext_vector_type(4)))  unsigned int v4u;
typedef __attribute__((ext_vector_type(8)))  int v8i;
typedef __attribute__((ext_vector_type(4)))  int v4i;

#if __has_include(<hip/amd_detail/amd_gfx1250_TDM.h>)
#define TDM_6ARG 1
#endif

struct Frag {
    union { v16bf bf; unsigned u[8]; };
};

__device__ inline unsigned short f2bf(float f) {
    unsigned u = __builtin_bit_cast(unsigned, f);
    unsigned r = (u + 0x7FFFu + ((u >> 16) & 1u)) >> 16;   // round-to-nearest-even
    return (unsigned short)r;
}
__device__ inline float bf2f(unsigned short h) {
    unsigned u = ((unsigned)h) << 16;
    return __builtin_bit_cast(float, u);
}
__device__ inline float sigmoidf_(float x) { return 1.0f / (1.0f + __expf(-x)); }

// Load a 16x32 bf16 A/B fragment per ISA layout:
//   lane: row (A: M, B: N) = lane&15 (passed in as `row`), khalf = lane>>4
//   VGPR v holds K = base, base+1 where base = (v<4?0:16) + khalf*8 + 2*(v&3)
__device__ inline void load_frag(Frag& f, const unsigned short* __restrict__ base,
                                 int row, int ld, int k0, int lane) {
    int khalf = (lane >> 4) & 1;
#pragma unroll
    for (int v = 0; v < 8; ++v) {
        int kb = ((v < 4) ? 0 : 16) + khalf * 8 + (v & 3) * 2;
        f.u[v] = *reinterpret_cast<const unsigned*>(base + (size_t)row * ld + k0 + kb);
    }
}

// Same but for LDS-resident (generic pointer -> ds_load after addrspace infer)
__device__ inline void load_frag_lds(Frag& f, const unsigned short* base,
                                     int row, int ld, int k0, int lane) {
    int khalf = (lane >> 4) & 1;
#pragma unroll
    for (int v = 0; v < 8; ++v) {
        int kb = ((v < 4) ? 0 : 16) + khalf * 8 + (v & 3) * 2;
        f.u[v] = *reinterpret_cast<const unsigned*>(base + (size_t)row * ld + k0 + kb);
    }
}

__device__ inline v8f wmma_bf(const Frag& a, const Frag& b, v8f c) {
    return __builtin_amdgcn_wmma_f32_16x16x32_bf16(false, a.bf, false, b.bf,
                                                   (short)0, c, false, false);
}

// ---------------------------------------------------------------------------
// Prologue: fp32 -> bf16 weight/emb conversion, zero c buffers + barrier
// ---------------------------------------------------------------------------
__global__ __launch_bounds__(256) void prologue_k(
    const float* __restrict__ emb,
    const float* __restrict__ fx, const float* __restrict__ ix,
    const float* __restrict__ ox, const float* __restrict__ cx,
    const float* __restrict__ fh, const float* __restrict__ ih,
    const float* __restrict__ oh,
    unsigned short* __restrict__ embb,   // 128x128
    unsigned short* __restrict__ Wx,     // 4 x 256x128
    unsigned short* __restrict__ Wh,     // 3 x 256x256
    unsigned short* __restrict__ cT0, unsigned short* __restrict__ cT1,
    float* __restrict__ c32a, float* __restrict__ c32b,
    unsigned* __restrict__ bar)
{
    size_t i = (size_t)blockIdx.x * 256 + threadIdx.x;
    size_t stride = (size_t)gridDim.x * 256;
    for (size_t j = i; j < (size_t)Cd * Cd; j += stride) embb[j] = f2bf(emb[j]);
    for (size_t j = i; j < (size_t)Hd * Cd; j += stride) {
        Wx[j]              = f2bf(fx[j]);
        Wx[32768 + j]      = f2bf(ix[j]);
        Wx[65536 + j]      = f2bf(ox[j]);
        Wx[98304 + j]      = f2bf(cx[j]);
    }
    for (size_t j = i; j < (size_t)Hd * Hd; j += stride) {
        Wh[j]               = f2bf(fh[j]);
        Wh[65536 + j]       = f2bf(ih[j]);
        Wh[131072 + j]      = f2bf(oh[j]);
        cT0[j] = 0; cT1[j] = 0;
        c32a[j] = 0.0f; c32b[j] = 0.0f;
    }
    if (i < 64) bar[i] = 0;
}

// ---------------------------------------------------------------------------
// Kernel 1: gate pre-activations  Xg[s,h,b] = sum_c Wg[h,c] * emb[x[b,s],c]
//   grid (1023, 4), 256 threads (8 waves). WG: step s, 64 b-columns, all 256 h.
//   wave w: btile = w&3 (16 cols), hhalf = w>>2 (128 rows = 8 tiles)
// ---------------------------------------------------------------------------
__global__ __launch_bounds__(256) void gates_k(
    const int* __restrict__ x,
    const unsigned short* __restrict__ embb,  // [tok][c] bf16
    const unsigned short* __restrict__ Wx,    // 4 x [h][c] bf16
    unsigned short* __restrict__ XG)          // 4 x [s][h][b] bf16
{
    const int s     = blockIdx.x;
    const int b0    = blockIdx.y * 64;
    const int lane  = threadIdx.x & 31;
    const int w     = threadIdx.x >> 5;
    const int btile = w & 3;
    const int hhalf = w >> 2;
    const int n     = lane & 15;
    const int half  = lane >> 4;
    const int b     = b0 + btile * 16 + n;
    const int tok   = x[(size_t)b * SEQ + s];
    const size_t GSZ = (size_t)NSTEP * Hd * Bd;
    const v8f vzero = {0.f, 0.f, 0.f, 0.f, 0.f, 0.f, 0.f, 0.f};

    // Hoist all 4 K-chunks of the gathered-embedding B fragments: shared by
    // all 4 gates (saves 3/4 of emb reads).
    Frag Bf[4];
#pragma unroll
    for (int kc = 0; kc < Cd / 32; ++kc)
        load_frag(Bf[kc], embb, tok, Cd, kc * 32, lane);

    for (int g = 0; g < 4; ++g) {
        v8f acc[8];
#pragma unroll
        for (int ht = 0; ht < 8; ++ht) acc[ht] = vzero;

#pragma unroll
        for (int kc = 0; kc < Cd / 32; ++kc) {
#pragma unroll
            for (int ht = 0; ht < 8; ++ht) {
                Frag Af;
                load_frag(Af, Wx + (size_t)g * Hd * Cd,
                          hhalf * 128 + ht * 16 + n, Cd, kc * 32, lane);
                acc[ht] = wmma_bf(Af, Bf[kc], acc[ht]);
            }
        }
        unsigned short* dst = XG + (size_t)g * GSZ + (size_t)s * Hd * Bd;
#pragma unroll
        for (int ht = 0; ht < 8; ++ht) {
#pragma unroll
            for (int r = 0; r < 8; ++r) {
                int h = hhalf * 128 + ht * 16 + r + half * 8;
                dst[(size_t)h * Bd + b] = f2bf(acc[ht][r]);
            }
        }
    }
}

// ---------------------------------------------------------------------------
// Kernel 2: persistent recurrence. 16 WGs (4x4 tiles of 64x64 over HxB),
// one device-wide barrier per step, double-buffered c.
// Wh rows for this WG staged once into LDS via the Tensor Data Mover with
// hardware padding (544B row pitch) to kill LDS bank conflicts.
// ---------------------------------------------------------------------------
__device__ inline void grid_barrier(unsigned* cnt, unsigned* gen, int step) {
    __threadfence();
    __syncthreads();
    if (threadIdx.x == 0) {
        unsigned target = 16u * (unsigned)(step + 1);
        unsigned prev = atomicAdd(cnt, 1u);
        if (prev == target - 1u) {
            atomicExch(gen, (unsigned)(step + 1));
        } else {
            while (atomicAdd(gen, 0u) < (unsigned)(step + 1)) {
                __builtin_amdgcn_s_sleep(1);
            }
        }
    }
    __syncthreads();
    __threadfence();
}

__global__ __launch_bounds__(256) void lstm_rec_k(
    const unsigned short* __restrict__ XG,   // 4 x [s][h][b] bf16
    const unsigned short* __restrict__ Wh,   // 3 x [h][h'] bf16 (f,i,o)
    const float* __restrict__ fb, const float* __restrict__ ib,
    const float* __restrict__ ob, const float* __restrict__ cb,
    unsigned short* __restrict__ cT0, unsigned short* __restrict__ cT1, // [b][h'] bf16
    float* __restrict__ c32a, float* __restrict__ c32b,                 // [h][b] f32
    float* __restrict__ hfin,                                           // [h][b] f32
    unsigned* __restrict__ bar_cnt, unsigned* __restrict__ bar_gen)
{
    const int wg   = blockIdx.x;             // 0..15
    const int h0   = (wg & 3) * 64;
    const int b0   = (wg >> 2) * 64;
    const int lane = threadIdx.x & 31;
    const int w    = threadIdx.x >> 5;
    const int mt   = w & 3;                  // 16-row tile within 64
    const int ntb  = (w >> 2) * 2;           // two 16-col tiles
    const int n    = lane & 15;
    const int half = lane >> 4;
    const size_t GSZ = (size_t)NSTEP * Hd * Bd;
    const v8f vzero = {0.f, 0.f, 0.f, 0.f, 0.f, 0.f, 0.f, 0.f};

    __shared__ float sb[4][Hd];
    __shared__ unsigned short shWh[3 * 64 * WH_PITCH];   // ~104 KB, padded rows

    if (threadIdx.x < Hd) {
        sb[0][threadIdx.x] = fb[threadIdx.x];
        sb[1][threadIdx.x] = ib[threadIdx.x];
        sb[2][threadIdx.x] = ob[threadIdx.x];
        sb[3][threadIdx.x] = cb[threadIdx.x];
    }

#if __has_builtin(__builtin_amdgcn_tensor_load_to_lds)
    // --- TDM: one-shot DMA of this WG's 64 rows of fh/ih/oh into LDS ------
    if (w == 0) {
#pragma unroll
        for (int g = 0; g < 3; ++g) {
            unsigned long long ga = (unsigned long long)(size_t)
                (Wh + (size_t)g * Hd * Hd + (size_t)h0 * Hd);
            unsigned lds = (unsigned)(size_t)(&shWh[(size_t)g * 64 * WH_PITCH]);
            // D# group 0: count=1 | lds_addr | global_addr[56:0] | type=2
            v4u g0 = { 1u,
                       lds,
                       (unsigned)(ga & 0xffffffffu),
                       (unsigned)((ga >> 32) & 0x01ffffffu) | (2u << 30) };
            // D# group 1: data_size=2B, pad_enable, pad_interval=128dw(512B),
            // pad_amount=8dw(32B); tensor 256x64, tile 256x64, stride0=256.
            v8i g1 = { (int)((1u << 16) | (1u << 20) | (6u << 22) | (7u << 25)),
                       (int)(256u << 16),              // tensor_dim0.lo
                       (int)(64u << 16),               // tensor_dim0.hi | tensor_dim1.lo
                       (int)(256u << 16),              // tensor_dim1.hi | tile_dim0
                       (int)(64u),                     // tile_dim1 | tile_dim2=0
                       (int)(256),                     // tensor_dim0_stride.lo
                       0, 0 };
            v4i gz = {0, 0, 0, 0};
#ifdef TDM_6ARG
            v8i gz8 = {0, 0, 0, 0, 0, 0, 0, 0};
            __builtin_amdgcn_tensor_load_to_lds(g0, g1, gz, gz, gz8, 0);
#else
            __builtin_amdgcn_tensor_load_to_lds(g0, g1, gz, gz, 0);
#endif
        }
        __builtin_amdgcn_s_wait_tensorcnt(0);
    }
#else
    // Fallback: manual global->LDS copy with same padded layout.
    for (int idx = threadIdx.x; idx < 3 * 64 * 128; idx += 256) {
        int g = idx / (64 * 128), rem = idx % (64 * 128);
        int row = rem / 128, kk = rem % 128;   // kk indexes u32 pairs
        const unsigned* src = reinterpret_cast<const unsigned*>(
            Wh + (size_t)g * Hd * Hd + (size_t)(h0 + row) * Hd);
        unsigned* dstp = reinterpret_cast<unsigned*>(
            &shWh[((size_t)g * 64 + row) * WH_PITCH]);
        dstp[kk] = src[kk];
    }
#endif
    __syncthreads();

    for (int t = 0; t < NSTEP; ++t) {
        const unsigned short* cTp = (t & 1) ? cT1 : cT0;
        unsigned short*       cTn = (t & 1) ? cT0 : cT1;
        const float*          cp32 = (t & 1) ? c32b : c32a;
        float*                cn32 = (t & 1) ? c32a : c32b;

        // Prefetch next step's gate tiles while this step computes.
        if (t + 1 < NSTEP) {
            size_t nb = (size_t)(t + 1) * Hd * Bd
                      + (size_t)(h0 + (threadIdx.x >> 2)) * Bd
                      + b0 + (threadIdx.x & 3) * 16;
#pragma unroll
            for (int g = 0; g < 4; ++g)
                __builtin_prefetch(XG + (size_t)g * GSZ + nb, 0, 1);
        }

        v8f acc[3][2];
#pragma unroll
        for (int g = 0; g < 3; ++g) { acc[g][0] = vzero; acc[g][1] = vzero; }

        for (int kc = 0; kc < Hd / 32; ++kc) {
            Frag B0, B1;
            load_frag(B0, cTp, b0 + ntb * 16 + n,       Hd, kc * 32, lane);
            load_frag(B1, cTp, b0 + (ntb + 1) * 16 + n, Hd, kc * 32, lane);
#pragma unroll
            for (int g = 0; g < 3; ++g) {
                Frag Af;
                load_frag_lds(Af, shWh + (size_t)g * 64 * WH_PITCH,
                              mt * 16 + n, WH_PITCH, kc * 32, lane);
                acc[g][0] = wmma_bf(Af, B0, acc[g][0]);
                acc[g][1] = wmma_bf(Af, B1, acc[g][1]);
            }
        }

        const size_t tbase = (size_t)t * Hd * Bd;
#pragma unroll
        for (int j = 0; j < 2; ++j) {
            const int b = b0 + (ntb + j) * 16 + n;
            float cvals[8];
#pragma unroll
            for (int r = 0; r < 8; ++r) {
                const int h = h0 + mt * 16 + r + half * 8;
                const size_t idx = tbase + (size_t)h * Bd + b;
                float xf = bf2f(XG[0 * GSZ + idx]);
                float xi = bf2f(XG[1 * GSZ + idx]);
                float xo = bf2f(XG[2 * GSZ + idx]);
                float xc = bf2f(XG[3 * GSZ + idx]);
                float fg = sigmoidf_(acc[0][j][r] + xf + sb[0][h]);
                float ig = sigmoidf_(acc[1][j][r] + xi + sb[1][h]);
                float og = sigmoidf_(acc[2][j][r] + xo + sb[2][h]);
                float cs = sigmoidf_(xc + sb[3][h]);
                float cv = cp32[(size_t)h * Bd + b];
                float cn = cs * ig + cv * fg;
                cn32[(size_t)h * Bd + b] = cn;
                cvals[r] = cn;
                if (t == NSTEP - 1) hfin[(size_t)h * Bd + b] = tanhf(cn) * og;
            }
            // transposed bf16 copy for next step's B fragments (K contiguous)
#pragma unroll
            for (int r = 0; r < 8; r += 2) {
                const int h = h0 + mt * 16 + r + half * 8;
                unsigned pk = (unsigned)f2bf(cvals[r]) |
                              ((unsigned)f2bf(cvals[r + 1]) << 16);
                *reinterpret_cast<unsigned*>(cTn + (size_t)b * Hd + h) = pk;
            }
        }
        grid_barrier(bar_cnt, bar_gen, t);
    }
}

// ---------------------------------------------------------------------------
// Kernel 3: p = ph @ h_fin + pb ; softmax over batch axis ; out = y.T (B,C)
// ---------------------------------------------------------------------------
__global__ __launch_bounds__(256) void proj_softmax_k(
    const float* __restrict__ ph, const float* __restrict__ pb,
    const float* __restrict__ hfin, float* __restrict__ out)
{
    const int c = blockIdx.x;    // 0..127
    const int b = threadIdx.x;   // 0..255
    float acc = pb[c];
    for (int h = 0; h < Hd; ++h)
        acc += ph[(size_t)c * Hd + h] * hfin[(size_t)h * Bd + b];

    __shared__ float red[Bd];
    red[b] = acc;
    __syncthreads();
    for (int off = 128; off > 0; off >>= 1) {
        if (b < off) red[b] = fmaxf(red[b], red[b + off]);
        __syncthreads();
    }
    float mx = red[0];
    __syncthreads();
    float e = __expf(acc - mx);
    red[b] = e;
    __syncthreads();
    for (int off = 128; off > 0; off >>= 1) {
        if (b < off) red[b] += red[b + off];
        __syncthreads();
    }
    out[(size_t)b * Cd + c] = e / red[0];
}

// ---------------------------------------------------------------------------
extern "C" void kernel_launch(void* const* d_in, const int* in_sizes, int n_in,
                              void* d_out, int out_size, void* d_ws, size_t ws_size,
                              hipStream_t stream) {
    (void)in_sizes; (void)n_in; (void)out_size; (void)ws_size;
    const int*   x   = (const int*)  d_in[0];
    const float* emb = (const float*)d_in[1];
    const float* fx  = (const float*)d_in[2];
    const float* fh  = (const float*)d_in[3];
    const float* fb  = (const float*)d_in[4];
    const float* ix  = (const float*)d_in[5];
    const float* ih  = (const float*)d_in[6];
    const float* ib  = (const float*)d_in[7];
    const float* ox  = (const float*)d_in[8];
    const float* oh  = (const float*)d_in[9];
    const float* ob  = (const float*)d_in[10];
    const float* cx  = (const float*)d_in[11];
    const float* cb  = (const float*)d_in[12];
    const float* ph  = (const float*)d_in[13];
    const float* pb  = (const float*)d_in[14];
    float* out = (float*)d_out;

    char* ws = (char*)d_ws;
    size_t off = 0;
    auto take = [&](size_t bytes) {
        size_t o = off;
        off += (bytes + 255) & ~(size_t)255;
        return o;
    };
    const size_t GSZ = (size_t)NSTEP * Hd * Bd;           // elems per gate
    unsigned short* XG   = (unsigned short*)(ws + take(4 * GSZ * 2));
    unsigned short* embb = (unsigned short*)(ws + take((size_t)Cd * Cd * 2));
    unsigned short* Wx   = (unsigned short*)(ws + take((size_t)4 * Hd * Cd * 2));
    unsigned short* Wh   = (unsigned short*)(ws + take((size_t)3 * Hd * Hd * 2));
    unsigned short* cT0  = (unsigned short*)(ws + take((size_t)Hd * Bd * 2));
    unsigned short* cT1  = (unsigned short*)(ws + take((size_t)Hd * Bd * 2));
    float*          c32a = (float*)(ws + take((size_t)Hd * Bd * 4));
    float*          c32b = (float*)(ws + take((size_t)Hd * Bd * 4));
    float*          hfin = (float*)(ws + take((size_t)Hd * Bd * 4));
    unsigned*       bar  = (unsigned*)(ws + take(256));
    unsigned* bar_cnt = bar;
    unsigned* bar_gen = bar + 1;

    prologue_k<<<256, 256, 0, stream>>>(emb, fx, ix, ox, cx, fh, ih, oh,
                                        embb, Wx, Wh, cT0, cT1, c32a, c32b, bar);
    gates_k<<<dim3(NSTEP, 4, 1), 256, 0, stream>>>(x, embb, Wx, XG);
    lstm_rec_k<<<16, 256, 0, stream>>>(XG, Wh, fb, ib, ob, cb,
                                       cT0, cT1, c32a, c32b, hfin,
                                       bar_cnt, bar_gen);
    proj_softmax_k<<<Cd, Bd, 0, stream>>>(ph, pb, hfin, out);
}
